// InterpretableMultiHeadAttention_19198503813785
// MI455X (gfx1250) — compile-verified, matrix operand order checked by reading, not run on hardware
//
#include <hip/hip_runtime.h>
#include <stdint.h>

#define D_MODEL 1024
#define NHEADS  16
#define DK      64
#define SEQ     2048
#define BATCH   2

typedef __attribute__((ext_vector_type(16))) __bf16 v16bf;
typedef __attribute__((ext_vector_type(8)))  float  v8f;

union FragCvt { uint32_t u[8]; v16bf v; };

__device__ __forceinline__ uint32_t f2bf1(float f) {
    uint32_t x = __float_as_uint(f);
    return (x + 0x7FFFu + ((x >> 16) & 1u)) >> 16;   // round-to-nearest-even bf16
}
__device__ __forceinline__ uint32_t packbf(float lo, float hi) {
    return f2bf1(lo) | (f2bf1(hi) << 16);
}

// A/B fragment for v_wmma_f32_16x16x32_bf16, loaded from a row-major bf16 tile.
// ISA 16-bit A 16x32 layout: lanes 0-15 row M=lane, K chunks [0,8)+[16,24);
// lanes 16-31 row M=lane-16, K chunks [8,16)+[24,32). Two aligned 16B loads.
__device__ __forceinline__ v16bf load_frag_bf16(const uint16_t* base, int stride, int lane) {
    const int r  = lane & 15;
    const int hf = lane >> 4;
    const uint16_t* p = base + (size_t)r * stride + hf * 8;
    uint4 a = *(const uint4*)(p);
    uint4 b = *(const uint4*)(p + 16);
    FragCvt f;
    f.u[0] = a.x; f.u[1] = a.y; f.u[2] = a.z; f.u[3] = a.w;
    f.u[4] = b.x; f.u[5] = b.y; f.u[6] = b.z; f.u[7] = b.w;
    return f.v;
}

__device__ __forceinline__ v8f wmma_bf16(v16bf a, v16bf b, v8f c) {
    return __builtin_amdgcn_wmma_f32_16x16x32_bf16(false, a, false, b, (short)0, c, false, false);
}

// ---------------------------------------------------------------------------
// One-shot fp32 -> bf16 cast (vectorized b128 read / b64 write). Streams every
// operand exactly once so the GEMM hot loops are pure bf16 loads + WMMA.
// ---------------------------------------------------------------------------
__global__ void __launch_bounds__(256)
cast_f32_bf16(const float* __restrict__ src, uint16_t* __restrict__ dst, int n4) {
    const int idx = blockIdx.x * 256 + threadIdx.x;
    if (idx < n4) {
        float4 v = ((const float4*)src)[idx];
        uint2 p;
        p.x = packbf(v.x, v.y);
        p.y = packbf(v.z, v.w);
        ((uint2*)dst)[idx] = p;
    }
}

// ---------------------------------------------------------------------------
// Projection GEMM: C[M=4096][N=1024] = A(bf16, MxK) @ W(bf16, NxK)^T, K=1024.
// Block = 8 waves; wave computes a 16x64 strip (4 n-tiles, A-frag reused).
// MODE 0: store bf16 row-major (Q, K). MODE 1: store bf16 transposed per head
// as Vt[b][h][d][s]. MODE 2: store f32 row-major (final output projection).
// ---------------------------------------------------------------------------
template <int MODE>
__global__ void __launch_bounds__(256)
proj_gemm(const uint16_t* __restrict__ A, const uint16_t* __restrict__ W, void* __restrict__ Out) {
    const int lane = threadIdx.x & 31;
    const int wave = threadIdx.x >> 5;
    const int m0 = blockIdx.x * 128 + wave * 16;
    const int n0 = blockIdx.y * 64;

    v8f acc[4] = {};
    for (int kk = 0; kk < D_MODEL; kk += 32) {
        v16bf af = load_frag_bf16(A + (size_t)m0 * D_MODEL + kk, D_MODEL, lane);
#pragma unroll
        for (int t = 0; t < 4; ++t) {
            v16bf bf = load_frag_bf16(W + (size_t)(n0 + t * 16) * D_MODEL + kk, D_MODEL, lane);
            acc[t] = wmma_bf16(af, bf, acc[t]);
        }
    }

    const int nl = lane & 15;
    const int mb = (lane >> 4) * 8;   // C layout: lane holds col nl, rows mb..mb+7

    if (MODE == 0) {
        uint16_t* O = (uint16_t*)Out;
#pragma unroll
        for (int t = 0; t < 4; ++t)
#pragma unroll
            for (int i = 0; i < 8; ++i)
                O[(size_t)(m0 + mb + i) * D_MODEL + n0 + t * 16 + nl] = (uint16_t)f2bf1(acc[t][i]);
    } else if (MODE == 1) {
        uint16_t* O = (uint16_t*)Out;          // Vt[b][h][d][s]
        const int m  = m0 + mb;                // rows -> consecutive s: one b128 store
        const int bb = m >> 11;
        const int s  = m & (SEQ - 1);
#pragma unroll
        for (int t = 0; t < 4; ++t) {
            const int n = n0 + t * 16 + nl;
            const int h = n >> 6, d = n & 63;
            uint4 pk;
            pk.x = packbf(acc[t][0], acc[t][1]);
            pk.y = packbf(acc[t][2], acc[t][3]);
            pk.z = packbf(acc[t][4], acc[t][5]);
            pk.w = packbf(acc[t][6], acc[t][7]);
            *(uint4*)(O + (((size_t)bb * NHEADS + h) * DK + d) * SEQ + s) = pk;
        }
    } else {
        float* O = (float*)Out;
#pragma unroll
        for (int t = 0; t < 4; ++t)
#pragma unroll
            for (int i = 0; i < 8; ++i)
                O[(size_t)(m0 + mb + i) * D_MODEL + n0 + t * 16 + nl] = acc[t][i];
    }
}

// ---------------------------------------------------------------------------
// Pass 1: per (b,h,q) row-max and sum-exp of scaled scores (two L2 sweeps).
// Wave owns 16 q-rows; block = 8 waves = 128 rows. Grid = B*H*16.
// ---------------------------------------------------------------------------
__global__ void __launch_bounds__(256)
attn_stats(const uint16_t* __restrict__ Qbf, const uint16_t* __restrict__ Kbf,
           float* __restrict__ stats) {
    const int lane = threadIdx.x & 31;
    const int wave = threadIdx.x >> 5;
    const int qc = blockIdx.x & 15;
    const int h  = (blockIdx.x >> 4) & 15;
    const int b  = blockIdx.x >> 8;
    const int q0 = qc * 128 + wave * 16;
    const int mb = (lane >> 4) * 8;
    const float scale = 0.125f;   // 1/sqrt(64)

    const uint16_t* Qrow = Qbf + (size_t)(b * SEQ + q0) * D_MODEL + h * DK;
    v16bf qf0 = load_frag_bf16(Qrow,      D_MODEL, lane);
    v16bf qf1 = load_frag_bf16(Qrow + 32, D_MODEL, lane);

    float mloc[8];
#pragma unroll
    for (int i = 0; i < 8; ++i) mloc[i] = -3.0e38f;

    for (int j0 = 0; j0 < SEQ; j0 += 16) {
        const uint16_t* Krow = Kbf + (size_t)(b * SEQ + j0) * D_MODEL + h * DK;
        v8f acc = {};
        acc = wmma_bf16(qf0, load_frag_bf16(Krow,      D_MODEL, lane), acc);
        acc = wmma_bf16(qf1, load_frag_bf16(Krow + 32, D_MODEL, lane), acc);
#pragma unroll
        for (int i = 0; i < 8; ++i) mloc[i] = fmaxf(mloc[i], acc[i] * scale);
    }
    // reduce max across the 16 lanes holding each row (stays within half-wave)
#pragma unroll
    for (int i = 0; i < 8; ++i)
#pragma unroll
        for (int off = 1; off < 16; off <<= 1)
            mloc[i] = fmaxf(mloc[i], __shfl_xor(mloc[i], off, 32));

    float sloc[8] = {};
    for (int j0 = 0; j0 < SEQ; j0 += 16) {
        const uint16_t* Krow = Kbf + (size_t)(b * SEQ + j0) * D_MODEL + h * DK;
        v8f acc = {};
        acc = wmma_bf16(qf0, load_frag_bf16(Krow,      D_MODEL, lane), acc);
        acc = wmma_bf16(qf1, load_frag_bf16(Krow + 32, D_MODEL, lane), acc);
#pragma unroll
        for (int i = 0; i < 8; ++i) sloc[i] += __expf(acc[i] * scale - mloc[i]);
    }
#pragma unroll
    for (int i = 0; i < 8; ++i)
#pragma unroll
        for (int off = 1; off < 16; off <<= 1)
            sloc[i] += __shfl_xor(sloc[i], off, 32);

    if ((lane & 15) == 0) {
        float* mx = stats + (size_t)(b * NHEADS + h) * SEQ;
        float* sm = mx + (size_t)BATCH * NHEADS * SEQ;
        const int qb = q0 + mb;
#pragma unroll
        for (int i = 0; i < 8; ++i) { mx[qb + i] = mloc[i]; sm[qb + i] = sloc[i]; }
    }
}

// ---------------------------------------------------------------------------
// Pass 2a: head outputs O = softmax(QK^T/8) @ V, exact via precomputed stats.
// P tile staged bf16 in padded LDS (stride 40 -> conflict-free b128 frag reads)
// and re-fed as the WMMA A operand for P@V. Output stored bf16 so the final
// projection also runs pure-bf16. Grid = B*H*16, block = 8 waves.
// ---------------------------------------------------------------------------
__global__ void __launch_bounds__(256)
attn_out(const uint16_t* __restrict__ Qbf, const uint16_t* __restrict__ Kbf,
         const uint16_t* __restrict__ Vt, const float* __restrict__ stats,
         uint16_t* __restrict__ head_out) {
    __shared__ uint16_t Pstage[8][16][40];
    const int lane = threadIdx.x & 31;
    const int wave = threadIdx.x >> 5;
    const int qc = blockIdx.x & 15;
    const int h  = (blockIdx.x >> 4) & 15;
    const int b  = blockIdx.x >> 8;
    const int q0 = qc * 128 + wave * 16;
    const int nl = lane & 15;
    const int mb = (lane >> 4) * 8;

    const uint16_t* Qrow = Qbf + (size_t)(b * SEQ + q0) * D_MODEL + h * DK;
    v16bf qf0 = load_frag_bf16(Qrow,      D_MODEL, lane);
    v16bf qf1 = load_frag_bf16(Qrow + 32, D_MODEL, lane);

    float rowm[8], rowinv[8];
    {
        const float* mx = stats + (size_t)(b * NHEADS + h) * SEQ;
        const float* sm = mx + (size_t)BATCH * NHEADS * SEQ;
#pragma unroll
        for (int i = 0; i < 8; ++i) {
            rowm[i]   = mx[q0 + mb + i];
            rowinv[i] = 1.0f / sm[q0 + mb + i];
        }
    }

    uint16_t* myP = &Pstage[wave][0][0];
    const uint16_t* Vbase = Vt + (size_t)(b * NHEADS + h) * DK * SEQ;
    v8f oacc[4] = {};

    for (int j0 = 0; j0 < SEQ; j0 += 32) {
#pragma unroll
        for (int jt = 0; jt < 32; jt += 16) {
            const uint16_t* Krow = Kbf + (size_t)(b * SEQ + j0 + jt) * D_MODEL + h * DK;
            v8f acc = {};
            acc = wmma_bf16(qf0, load_frag_bf16(Krow,      D_MODEL, lane), acc);
            acc = wmma_bf16(qf1, load_frag_bf16(Krow + 32, D_MODEL, lane), acc);
#pragma unroll
            for (int i = 0; i < 8; ++i) {
                float p = __expf(acc[i] * 0.125f - rowm[i]) * rowinv[i];
                myP[(mb + i) * 40 + jt + nl] = (uint16_t)f2bf1(p);
            }
        }
        __syncthreads();   // all waves iterate uniformly; per-wave LDS tiles
        v16bf pf = load_frag_bf16(myP, 40, lane);
#pragma unroll
        for (int t = 0; t < 4; ++t) {
            v16bf vf = load_frag_bf16(Vbase + (size_t)(t * 16) * SEQ + j0, SEQ, lane);
            oacc[t] = wmma_bf16(pf, vf, oacc[t]);
        }
        __syncthreads();
    }

#pragma unroll
    for (int t = 0; t < 4; ++t)
#pragma unroll
        for (int i = 0; i < 8; ++i)
            head_out[(size_t)(b * SEQ + q0 + mb + i) * D_MODEL + h * DK + t * 16 + nl] =
                (uint16_t)f2bf1(oacc[t][i]);
}

// ---------------------------------------------------------------------------
// Pass 2b: attention-map head mean. Wave owns one 16x16 (q,k) tile and loops
// all 16 heads, accumulating in registers; single write -> no atomics.
// Grid = B*(SEQ/16)*(SEQ/16)/8, block = 8 waves.
// ---------------------------------------------------------------------------
__global__ void __launch_bounds__(256)
attn_mean(const uint16_t* __restrict__ Qbf, const uint16_t* __restrict__ Kbf,
          const float* __restrict__ stats, float* __restrict__ attnw) {
    const int lane = threadIdx.x & 31;
    const int wave = threadIdx.x >> 5;
    const int flat = blockIdx.x * 8 + wave;
    const int jt = flat & 127;
    const int qt = (flat >> 7) & 127;
    const int b  = flat >> 14;
    const int q0 = qt * 16, j0 = jt * 16;
    const int nl = lane & 15;
    const int mb = (lane >> 4) * 8;

    float macc[8] = {};
    for (int h = 0; h < NHEADS; ++h) {
        const uint16_t* Qrow = Qbf + (size_t)(b * SEQ + q0) * D_MODEL + h * DK;
        const uint16_t* Krow = Kbf + (size_t)(b * SEQ + j0) * D_MODEL + h * DK;
        v8f acc = {};
        acc = wmma_bf16(load_frag_bf16(Qrow,      D_MODEL, lane),
                        load_frag_bf16(Krow,      D_MODEL, lane), acc);
        acc = wmma_bf16(load_frag_bf16(Qrow + 32, D_MODEL, lane),
                        load_frag_bf16(Krow + 32, D_MODEL, lane), acc);
        const float* mx = stats + (size_t)(b * NHEADS + h) * SEQ;
        const float* sm = mx + (size_t)BATCH * NHEADS * SEQ;
#pragma unroll
        for (int i = 0; i < 8; ++i)
            macc[i] += __expf(acc[i] * 0.125f - mx[q0 + mb + i]) / sm[q0 + mb + i];
    }
    const float invH = 1.0f / (float)NHEADS;
#pragma unroll
    for (int i = 0; i < 8; ++i)
        attnw[((size_t)b * SEQ + q0 + mb + i) * SEQ + j0 + nl] = macc[i] * invH;
}

// ---------------------------------------------------------------------------
extern "C" void kernel_launch(void* const* d_in, const int* in_sizes, int n_in,
                              void* d_out, int out_size, void* d_ws, size_t ws_size,
                              hipStream_t stream) {
    (void)in_sizes; (void)n_in; (void)out_size; (void)ws_size;
    const float* query = (const float*)d_in[0];
    const float* key_  = (const float*)d_in[1];
    const float* value = (const float*)d_in[2];
    const float* W_q   = (const float*)d_in[3];
    const float* W_k   = (const float*)d_in[4];
    const float* W_v   = (const float*)d_in[5];
    const float* W_o   = (const float*)d_in[6];

    const size_t NW = (size_t)D_MODEL * D_MODEL;     // 1M elems per weight
    const size_t NX = (size_t)BATCH * SEQ * D_MODEL; // 4M elems per activation

    // workspace layout (~56.5 MB)
    uint16_t* Wq_bf = (uint16_t*)d_ws;               // 4 x 2 MB weights
    uint16_t* Wk_bf = Wq_bf + NW;
    uint16_t* Wv_bf = Wk_bf + NW;
    uint16_t* Wo_bf = Wv_bf + NW;
    uint16_t* Xq_bf = Wo_bf + NW;                    // 3 x 8 MB cast activations
    uint16_t* Xk_bf = Xq_bf + NX;
    uint16_t* Xv_bf = Xk_bf + NX;
    uint16_t* Qbf   = Xv_bf + NX;                    // 3 x 8 MB projected
    uint16_t* Kbf   = Qbf + NX;
    uint16_t* Vt    = Kbf + NX;                      // [b][h][d][s]
    float* stats    = (float*)(Vt + NX);             // 0.5 MB (max + sum planes)
    uint16_t* head_bf = Xq_bf;  // alias: Xq dead after Q projection (stream-ordered)

    float* out   = (float*)d_out;                    // (B,S,D_MODEL)
    float* attnw = out + NX;                         // (B,S,S)

    const dim3 blk(256);
    const dim3 gproj(BATCH * SEQ / 128, D_MODEL / 64);

    // one-shot bf16 casts (each element converted exactly once)
    cast_f32_bf16<<<dim3((unsigned)(NW / 4 / 256)), blk, 0, stream>>>(W_q, Wq_bf, (int)(NW / 4));
    cast_f32_bf16<<<dim3((unsigned)(NW / 4 / 256)), blk, 0, stream>>>(W_k, Wk_bf, (int)(NW / 4));
    cast_f32_bf16<<<dim3((unsigned)(NW / 4 / 256)), blk, 0, stream>>>(W_v, Wv_bf, (int)(NW / 4));
    cast_f32_bf16<<<dim3((unsigned)(NW / 4 / 256)), blk, 0, stream>>>(W_o, Wo_bf, (int)(NW / 4));
    cast_f32_bf16<<<dim3((unsigned)(NX / 4 / 256)), blk, 0, stream>>>(query, Xq_bf, (int)(NX / 4));
    cast_f32_bf16<<<dim3((unsigned)(NX / 4 / 256)), blk, 0, stream>>>(key_,  Xk_bf, (int)(NX / 4));
    cast_f32_bf16<<<dim3((unsigned)(NX / 4 / 256)), blk, 0, stream>>>(value, Xv_bf, (int)(NX / 4));

    // projections (pure bf16 operands)
    proj_gemm<0><<<gproj, blk, 0, stream>>>(Xq_bf, Wq_bf, (void*)Qbf);
    proj_gemm<0><<<gproj, blk, 0, stream>>>(Xk_bf, Wk_bf, (void*)Kbf);
    proj_gemm<1><<<gproj, blk, 0, stream>>>(Xv_bf, Wv_bf, (void*)Vt);

    // attention
    attn_stats<<<dim3(BATCH * NHEADS * 16), blk, 0, stream>>>(Qbf, Kbf, stats);
    attn_out  <<<dim3(BATCH * NHEADS * 16), blk, 0, stream>>>(Qbf, Kbf, Vt, stats, head_bf);
    attn_mean <<<dim3(BATCH * (SEQ / 16) * (SEQ / 16) / 8), blk, 0, stream>>>(Qbf, Kbf, stats, attnw);

    // output projection
    proj_gemm<2><<<gproj, blk, 0, stream>>>(head_bf, Wo_bf, d_out);
}
